// TransformerLayer_75849122447780
// MI455X (gfx1250) — matrix operand order repaired and checked by reading, not compiled, not run on hardware
//
#include <hip/hip_runtime.h>

// Problem constants (B,N,D,H from the reference)
#define B_  4
#define N_  2048
#define D_  1024
#define H_  16
#define DH_ 64
#define HB_ (H_ * B_)   // 64 head-batches
#define M_  (B_ * N_)   // 8192 rows for QKV GEMM

typedef __bf16 bf16;
typedef __attribute__((ext_vector_type(16))) __bf16 v16bf;
typedef __attribute__((ext_vector_type(8)))  __bf16 v8bf;
typedef __attribute__((ext_vector_type(8)))  float  v8f;

union Frag16 { v16bf v; v8bf h[2]; };

__device__ __forceinline__ v8f wmma_bf16(v16bf a, v16bf b, v8f c) {
  return __builtin_amdgcn_wmma_f32_16x16x32_bf16(false, a, false, b,
                                                 (short)0, c, false, false);
}

// ---- CDNA5 async copy: global -> LDS, 16B per lane, tracked by ASYNCcnt ----
// GVS mode: uniform 64-bit SGPR base + per-lane signed 32-bit byte offset.
__device__ __forceinline__ void async_ld_b128(uint32_t lds_addr,
                                              uint32_t gvoff_bytes,
                                              const void* sbase) {
  asm volatile("global_load_async_to_lds_b128 %0, %1, %2"
               :: "v"(lds_addr), "v"(gvoff_bytes), "s"(sbase)
               : "memory");
}
#define S_WAIT_ASYNC(n) asm volatile("s_wait_asynccnt " n ::: "memory")

// Reductions across the 16-lane halves of a wave32 (rows of the WMMA C layout
// live across lanes 0-15 / 16-31; xor masks <=8 never cross the halves).
__device__ __forceinline__ float half16_max(float v) {
#pragma unroll
  for (int m = 1; m <= 8; m <<= 1) v = fmaxf(v, __shfl_xor(v, m, 32));
  return v;
}
__device__ __forceinline__ float half16_sum(float v) {
#pragma unroll
  for (int m = 1; m <= 8; m <<= 1) v += __shfl_xor(v, m, 32);
  return v;
}

// ---------------------------------------------------------------------------
// Kernel 1: x = data + PositionalEncoding(N,D), emitted as bf16 [M_, D_].
// ---------------------------------------------------------------------------
__global__ void pe_add_kernel(const float* __restrict__ data,
                              bf16* __restrict__ xpe) {
  int idx = blockIdx.x * blockDim.x + threadIdx.x;   // over N_*D_
  int pos = idx >> 10;            // / D_
  int i   = idx & (D_ - 1);
  float fi = (float)(i & ~1);
  float denom_i = __powf(10000.f, 2.f * fi / (float)D_);
  float a = (float)pos / denom_i;
  float base = (i & 1) ? __cosf(a) : __sinf(a);
  float fp = (float)(pos & ~1);
  float denom_p = __powf(10000.f, 2.f * fp / (float)N_);
  float c = (float)i / denom_p;
  float mult = (pos & 1) ? __cosf(c) : __sinf(c);
  float pe = base * mult;
#pragma unroll
  for (int b = 0; b < B_; ++b) {
    int off = (b * N_ + pos) * D_ + i;
    xpe[off] = (bf16)(data[off] + pe);
  }
}

// ---------------------------------------------------------------------------
// Kernel 2: Wt[n][k] = (bf16)W[k][n]  (so B fragments are contiguous rows)
// ---------------------------------------------------------------------------
__global__ void wtrans_kernel(const float* __restrict__ W,
                              bf16* __restrict__ Wt) {
  int idx = blockIdx.x * blockDim.x + threadIdx.x;   // over D_*D_
  int n = idx >> 10;
  int k = idx & (D_ - 1);
  Wt[n * D_ + k] = (bf16)W[k * D_ + n];
}

// ---------------------------------------------------------------------------
// Kernel 3: out = X @ W + b, WMMA bf16 / f32 accumulate.
// Block = 8 waves, 128 rows x 64 cols; the shared Wt tile (64x32 bf16 = 4KB)
// is staged in LDS via async-to-LDS, software-pipelined over two statically
// indexed buffers (1 async instr / wave / k-slab).
// ---------------------------------------------------------------------------
__global__ __launch_bounds__(256) void qkv_gemm_kernel(
    const bf16* __restrict__ X, const bf16* __restrict__ Wt,
    const float* __restrict__ bias, bf16* __restrict__ out, int vtrans) {
  __shared__ __align__(64) bf16 wbuf[2][64 * 32];
  const int t    = threadIdx.x;
  const int wid  = t >> 5;
  const int lane = t & 31;
  const int lm   = lane & 15;
  const int hi   = (lane >> 4) & 1;
  const int rg   = blockIdx.x >> 4;
  const int cg   = blockIdx.x & 15;
  const int m0   = (rg * 8 + wid) * 16;
  const int n0   = cg * 64;

  // Staging map: thread t copies one 16B chunk (8 bf16) of the 64x32 tile.
  const int srow = t >> 2;           // 0..63 : output column within group
  const int scol = (t & 3) * 8;      // 0,8,16,24 : k element within slab
  const uint32_t gbase = (uint32_t)(n0 + srow) * D_ + scol;   // elements
  uint32_t ldst[2];
  ldst[0] = (uint32_t)(size_t)&wbuf[0][srow * 32 + scol];
  ldst[1] = (uint32_t)(size_t)&wbuf[1][srow * 32 + scol];

  v8f acc[4] = {};
  const bf16* xrow = X + (size_t)(m0 + lm) * D_;

  auto prefetch = [&](int b, int kb) {
    async_ld_b128(ldst[b], (gbase + (uint32_t)kb) * 2u, Wt);
  };
  auto compute = [&](const bf16* wb, int kb) {
    Frag16 a;
    a.h[0] = *(const v8bf*)(xrow + kb + hi * 16);
    a.h[1] = *(const v8bf*)(xrow + kb + 8 + hi * 16);
    v16bf b0 = *(const v16bf*)&wb[(0 * 16 + lm) * 32 + hi * 16];
    v16bf b1 = *(const v16bf*)&wb[(1 * 16 + lm) * 32 + hi * 16];
    v16bf b2 = *(const v16bf*)&wb[(2 * 16 + lm) * 32 + hi * 16];
    v16bf b3 = *(const v16bf*)&wb[(3 * 16 + lm) * 32 + hi * 16];
    acc[0] = wmma_bf16(a.v, b0, acc[0]);
    acc[1] = wmma_bf16(a.v, b1, acc[1]);
    acc[2] = wmma_bf16(a.v, b2, acc[2]);
    acc[3] = wmma_bf16(a.v, b3, acc[3]);
  };

  prefetch(0, 0);
  prefetch(1, 32);
#pragma unroll 1
  for (int kb = 0; kb < D_ - 64; kb += 64) {
    S_WAIT_ASYNC("1");  __syncthreads();   // slab kb ready, kb+32 in flight
    compute(wbuf[0], kb);
    __syncthreads();                       // reads done before overwrite
    prefetch(0, kb + 64);
    S_WAIT_ASYNC("1");  __syncthreads();   // slab kb+32 ready
    compute(wbuf[1], kb + 32);
    __syncthreads();
    prefetch(1, kb + 96);
  }
  // Tail: slabs D_-64 and D_-32 (no further prefetches).
  S_WAIT_ASYNC("1");  __syncthreads();
  compute(wbuf[0], D_ - 64);
  __syncthreads();
  S_WAIT_ASYNC("0");  __syncthreads();
  compute(wbuf[1], D_ - 32);

#pragma unroll
  for (int j = 0; j < 4; ++j) {
    int c  = n0 + j * 16 + lm;
    float bv = bias[c];
    int h  = c >> 6;       // head
    int dh = c & 63;
#pragma unroll
    for (int r = 0; r < 8; ++r) {
      int m    = m0 + r + 8 * hi;     // C layout: VGPR r, half hi -> row
      int bidx = m >> 11;             // / N_
      int n    = m & (N_ - 1);
      float val = acc[j][r] + bv;
      if (vtrans)
        out[((size_t)(h * B_ + bidx) * DH_ + dh) * N_ + n] = (bf16)val;
      else
        out[((size_t)(h * B_ + bidx) * N_ + n) * DH_ + dh] = (bf16)val;
    }
  }
}

// ---------------------------------------------------------------------------
// Kernel 4: flash attention. Block = 8 waves, one head-batch (hb uniform from
// blockIdx), 8 query tiles of 16 rows. K (32x64) and V (64x32) key-step tiles
// are staged in LDS via async-to-LDS, software-pipelined over two static
// buffers (2 async instr / wave / stage).
// Per 32-key stage: S = Q@K^T (4 WMMA), online softmax via shfl reductions,
// P relayout C->A through a per-wave LDS bounce, O += P@V (4 WMMA).
// ---------------------------------------------------------------------------
__global__ __launch_bounds__(256) void attn_kernel(
    const bf16* __restrict__ Q, const bf16* __restrict__ K,
    const bf16* __restrict__ Vt, float* __restrict__ out) {
  __shared__ __align__(64) bf16 kbuf[2][32 * 64];   // [key][dh]
  __shared__ __align__(64) bf16 vbuf[2][64 * 32];   // [dh][key]
  __shared__ __align__(64) bf16 pbuf[8][16 * 32];   // per-wave P bounce

  const int t    = threadIdx.x;
  const int wid  = t >> 5;
  const int lane = t & 31;
  const int lm   = lane & 15;
  const int hi   = (lane >> 4) & 1;
  const int hb   = blockIdx.x >> 4;                 // uniform per block
  const int q0   = ((blockIdx.x & 15) * 8 + wid) * 16;

  const bf16* Qb = Q  + (size_t)hb * (N_ * DH_);
  const bf16* Kb = K  + (size_t)hb * (N_ * DH_);
  const bf16* Vb = Vt + (size_t)hb * (DH_ * N_);
  bf16* wlds = &pbuf[wid][0];

  // Staging maps (one 16B chunk per thread per tile).
  const int krow = t >> 3, kcol = (t & 7) * 8;      // K tile 32x64
  const int vrow = t >> 2, vcol = (t & 3) * 8;      // V tile 64x32
  const uint32_t kgbase = (uint32_t)krow * DH_ + kcol;  // + kt*64 (elems)
  const uint32_t vgbase = (uint32_t)vrow * N_  + vcol;  // + kt     (elems)
  uint32_t kdst[2], vdst[2];
  kdst[0] = (uint32_t)(size_t)&kbuf[0][krow * 64 + kcol];
  kdst[1] = (uint32_t)(size_t)&kbuf[1][krow * 64 + kcol];
  vdst[0] = (uint32_t)(size_t)&vbuf[0][vrow * 32 + vcol];
  vdst[1] = (uint32_t)(size_t)&vbuf[1][vrow * 32 + vcol];

  // Q fragments (live in registers for the whole key sweep).
  Frag16 qf[2];
  {
    const bf16* qrow = Qb + (size_t)(q0 + lm) * DH_;
#pragma unroll
    for (int kb = 0; kb < 2; ++kb) {
      qf[kb].h[0] = *(const v8bf*)(qrow + kb * 32 + hi * 16);
      qf[kb].h[1] = *(const v8bf*)(qrow + kb * 32 + 8 + hi * 16);
    }
  }

  v8f   O[4] = {};
  float m8[8], l8[8];
#pragma unroll
  for (int r = 0; r < 8; ++r) { m8[r] = -3.0e38f; l8[r] = 0.f; }

  auto prefetch = [&](int b, int kt) {
    async_ld_b128(kdst[b], (kgbase + (uint32_t)kt * DH_) * 2u, Kb);
    async_ld_b128(vdst[b], (vgbase + (uint32_t)kt) * 2u, Vb);
  };

  auto stage = [&](const bf16* kb_lds, const bf16* vb_lds) {
    // ---- S = (Q @ K^T) * 1/sqrt(DH), two 16x16 tiles (32 keys) ----
    v16bf bk00 = *(const v16bf*)&kb_lds[(0 * 16 + lm) * 64 + 0 * 32 + hi * 16];
    v16bf bk01 = *(const v16bf*)&kb_lds[(0 * 16 + lm) * 64 + 1 * 32 + hi * 16];
    v16bf bk10 = *(const v16bf*)&kb_lds[(1 * 16 + lm) * 64 + 0 * 32 + hi * 16];
    v16bf bk11 = *(const v16bf*)&kb_lds[(1 * 16 + lm) * 64 + 1 * 32 + hi * 16];
    v8f s0 = {}, s1 = {};
    s0 = wmma_bf16(qf[0].v, bk00, s0);
    s0 = wmma_bf16(qf[1].v, bk01, s0);
    s1 = wmma_bf16(qf[0].v, bk10, s1);
    s1 = wmma_bf16(qf[1].v, bk11, s1);

    // ---- online softmax (row = r + 8*hi, spread across 16 lanes) ----
#pragma unroll
    for (int r = 0; r < 8; ++r) {
      float e0 = s0[r] * 0.125f;         // 1/sqrt(64)
      float e1 = s1[r] * 0.125f;
      float rowm = half16_max(fmaxf(e0, e1));
      float mnew = fmaxf(m8[r], rowm);
      float sc   = __expf(m8[r] - mnew); // 0 on first iteration
      float p0   = __expf(e0 - mnew);
      float p1   = __expf(e1 - mnew);
      l8[r] = l8[r] * sc + half16_sum(p0 + p1);
      m8[r] = mnew;
#pragma unroll
      for (int j = 0; j < 4; ++j) O[j][r] *= sc;
      wlds[(r + 8 * hi) * 32 + lm]      = (bf16)p0;
      wlds[(r + 8 * hi) * 32 + 16 + lm] = (bf16)p1;
    }
    asm volatile("s_wait_dscnt 0" ::: "memory");

    // ---- reload P in A-fragment layout (per-wave private bounce) ----
    Frag16 pa;
    pa.h[0] = *(const v8bf*)(wlds + lm * 32 + hi * 16);
    pa.h[1] = *(const v8bf*)(wlds + lm * 32 + 8 + hi * 16);

    // ---- O += P @ V, 4 dh tiles of 16 ----
    v16bf bv0 = *(const v16bf*)&vb_lds[(0 * 16 + lm) * 32 + hi * 16];
    v16bf bv1 = *(const v16bf*)&vb_lds[(1 * 16 + lm) * 32 + hi * 16];
    v16bf bv2 = *(const v16bf*)&vb_lds[(2 * 16 + lm) * 32 + hi * 16];
    v16bf bv3 = *(const v16bf*)&vb_lds[(3 * 16 + lm) * 32 + hi * 16];
    O[0] = wmma_bf16(pa.v, bv0, O[0]);
    O[1] = wmma_bf16(pa.v, bv1, O[1]);
    O[2] = wmma_bf16(pa.v, bv2, O[2]);
    O[3] = wmma_bf16(pa.v, bv3, O[3]);
  };

  prefetch(0, 0);
  prefetch(1, 32);
#pragma unroll 1
  for (int kt = 0; kt < N_ - 64; kt += 64) {
    S_WAIT_ASYNC("2");  __syncthreads();   // tile kt ready, kt+32 in flight
    stage(kbuf[0], vbuf[0]);
    __syncthreads();
    prefetch(0, kt + 64);
    S_WAIT_ASYNC("2");  __syncthreads();   // tile kt+32 ready
    stage(kbuf[1], vbuf[1]);
    __syncthreads();
    prefetch(1, kt + 96);
  }
  // Tail: key tiles N_-64 and N_-32.
  S_WAIT_ASYNC("2");  __syncthreads();
  stage(kbuf[0], vbuf[0]);
  __syncthreads();
  S_WAIT_ASYNC("0");  __syncthreads();
  stage(kbuf[1], vbuf[1]);

  // ---- normalize and write fp32 output [hb, n, dh] ----
#pragma unroll
  for (int j = 0; j < 4; ++j) {
#pragma unroll
    for (int r = 0; r < 8; ++r) {
      int row = q0 + r + 8 * hi;
      out[((size_t)hb * N_ + row) * DH_ + j * 16 + lm] = O[j][r] / l8[r];
    }
  }
}

// ---------------------------------------------------------------------------
extern "C" void kernel_launch(void* const* d_in, const int* in_sizes, int n_in,
                              void* d_out, int out_size, void* d_ws, size_t ws_size,
                              hipStream_t stream) {
  const float* data = (const float*)d_in[0];
  const float* Wq   = (const float*)d_in[1];
  const float* bq   = (const float*)d_in[2];
  const float* Wk   = (const float*)d_in[3];
  const float* bk   = (const float*)d_in[4];
  const float* Wv   = (const float*)d_in[5];
  const float* bv   = (const float*)d_in[6];
  float* out = (float*)d_out;

  // Workspace carve-up (~73.4 MB total, all 2^k-aligned offsets)
  char* p = (char*)d_ws;
  bf16* xpe = (bf16*)p;  p += (size_t)M_ * D_ * 2;        // 16 MB
  bf16* Wtq = (bf16*)p;  p += (size_t)D_ * D_ * 2;        //  2 MB
  bf16* Wtk = (bf16*)p;  p += (size_t)D_ * D_ * 2;
  bf16* Wtv = (bf16*)p;  p += (size_t)D_ * D_ * 2;
  bf16* Qm  = (bf16*)p;  p += (size_t)HB_ * N_ * DH_ * 2; // 16 MB
  bf16* Km  = (bf16*)p;  p += (size_t)HB_ * N_ * DH_ * 2;
  bf16* Vtm = (bf16*)p;                                   // V transposed [hb,dh,n]

  pe_add_kernel<<<(N_ * D_) / 256, 256, 0, stream>>>(data, xpe);
  wtrans_kernel<<<(D_ * D_) / 256, 256, 0, stream>>>(Wq, Wtq);
  wtrans_kernel<<<(D_ * D_) / 256, 256, 0, stream>>>(Wk, Wtk);
  wtrans_kernel<<<(D_ * D_) / 256, 256, 0, stream>>>(Wv, Wtv);

  qkv_gemm_kernel<<<1024, 256, 0, stream>>>(xpe, Wtq, bq, Qm, 0);
  qkv_gemm_kernel<<<1024, 256, 0, stream>>>(xpe, Wtk, bk, Km, 0);
  qkv_gemm_kernel<<<1024, 256, 0, stream>>>(xpe, Wtv, bv, Vtm, 1);

  attn_kernel<<<1024, 256, 0, stream>>>(Qm, Km, Vtm, out);
}
